// MPNN_32684701122624
// MI455X (gfx1250) — matrix operand order repaired and checked by reading, not compiled, not run on hardware
//
#include <hip/hip_runtime.h>
#include <math.h>

// ---------------------------------------------------------------------------
// MPNN forward for MI455X (gfx1250).
// Big matmuls (adj @ X, 10000x10000x32) via v_wmma_f32_16x16x32_bf16:
//   A (adjacency) read as f32, converted to bf16 in-register (native cvt).
//   B kept transposed as Xt[32][Kpad] bf16 so B-fragments are contiguous.
//   K split 4 ways per tile (deterministic tree reduction, no float atomics).
// Small MLP layers in plain f32 VALU.
// ---------------------------------------------------------------------------

#define NN      10000
#define KDIM    10000
#define KPAD    10016            // 313 chunks of 32
#define NCHUNK  313
#define NSPLIT  4
#define NTILES  625              // 10000 / 16

typedef __attribute__((ext_vector_type(16))) __bf16 v16bf;
typedef __attribute__((ext_vector_type(8)))  float  v8f;

__device__ __forceinline__ __bf16 f2bf(float f) {
    // clang fptrunc f32->bf16 is round-to-nearest-even; on gfx1250 this
    // selects the native convert (v_cvt_pk_bf16_f32 when pairs get fused).
    return static_cast<__bf16>(f);
}

// ---------------------------------------------------------------------------
// big_gemm: partial C[tile 16 x 32] = A[M x K](f32) @ Xt^T (bf16), K-split.
// part layout: [split][tile][ n(0..31) * 16 + mi(0..15) ]  (tile-col-major)
// ---------------------------------------------------------------------------
__global__ __launch_bounds__(32) void big_gemm_kernel(
    const float* __restrict__ A,       // [M x K] row-major f32
    const __bf16* __restrict__ Xt,     // [32 x KPAD] bf16
    float* __restrict__ part)          // [NSPLIT][NTILES][512]
{
    const int tile  = blockIdx.x;
    const int split = blockIdx.y;
    const int lane  = threadIdx.x;
    const int h     = lane >> 4;          // half select
    const int n     = lane & 15;          // col within 16 / row of A frag
    const int m     = tile * 16 + n;      // A row this lane feeds

    const int c0 = (split * NCHUNK) / NSPLIT;
    const int c1 = ((split + 1) * NCHUNK) / NSPLIT;

    const float*  Arow = A + (size_t)m * KDIM;
    const __bf16* B0   = Xt + (size_t)n * KPAD;
    const __bf16* B1   = Xt + (size_t)(n + 16) * KPAD;

    v8f acc0 = {};
    v8f acc1 = {};

#pragma unroll 2
    for (int c = c0; c < c1; ++c) {
        const int kb = c * 32;
        const int ka = kb + 8 * h;
        // A fragment: group0 = k[8h .. 8h+7], group1 = k[16+8h .. 16+8h+7]
        float4 a0 = *(const float4*)(Arow + ka);
        float4 a1 = *(const float4*)(Arow + ka + 4);
        float4 a2, a3;
        if (kb + 16 < KDIM) {             // only last chunk (K%32==16) masks
            a2 = *(const float4*)(Arow + ka + 16);
            a3 = *(const float4*)(Arow + ka + 20);
        } else {
            a2.x = a2.y = a2.z = a2.w = 0.0f;
            a3.x = a3.y = a3.z = a3.w = 0.0f;
        }
        v16bf av;
        av[0]  = f2bf(a0.x); av[1]  = f2bf(a0.y); av[2]  = f2bf(a0.z); av[3]  = f2bf(a0.w);
        av[4]  = f2bf(a1.x); av[5]  = f2bf(a1.y); av[6]  = f2bf(a1.z); av[7]  = f2bf(a1.w);
        av[8]  = f2bf(a2.x); av[9]  = f2bf(a2.y); av[10] = f2bf(a2.z); av[11] = f2bf(a2.w);
        av[12] = f2bf(a3.x); av[13] = f2bf(a3.y); av[14] = f2bf(a3.z); av[15] = f2bf(a3.w);

        // B fragment: group0 = k[8h .. 8h+7], group1 = k[16+8h .. 16+8h+7] (pad rows are 0)
        typedef __attribute__((ext_vector_type(8))) __bf16 v8bf;
        v16bf bv0 = __builtin_shufflevector(*(const v8bf*)(B0 + kb + 8 * h), *(const v8bf*)(B0 + kb + 16 + 8 * h), 0, 1, 2, 3, 4, 5, 6, 7, 8, 9, 10, 11, 12, 13, 14, 15);
        v16bf bv1 = __builtin_shufflevector(*(const v8bf*)(B1 + kb + 8 * h), *(const v8bf*)(B1 + kb + 16 + 8 * h), 0, 1, 2, 3, 4, 5, 6, 7, 8, 9, 10, 11, 12, 13, 14, 15);

        acc0 = __builtin_amdgcn_wmma_f32_16x16x32_bf16(false, av, false, bv0,
                                                       (short)0, acc0, false, false);
        acc1 = __builtin_amdgcn_wmma_f32_16x16x32_bf16(false, av, false, bv1,
                                                       (short)0, acc1, false, false);
    }

    // store: lane holds col n (+16), rows mi = 8h + r (r=0..7) -> contiguous
    float* base = part + ((size_t)split * NTILES + tile) * 512;
    float* p0 = base + (size_t)n * 16 + 8 * h;
    float* p1 = base + (size_t)(n + 16) * 16 + 8 * h;
    float4 s;
    s.x = acc0[0]; s.y = acc0[1]; s.z = acc0[2]; s.w = acc0[3]; *(float4*)(p0)     = s;
    s.x = acc0[4]; s.y = acc0[5]; s.z = acc0[6]; s.w = acc0[7]; *(float4*)(p0 + 4) = s;
    s.x = acc1[0]; s.y = acc1[1]; s.z = acc1[2]; s.w = acc1[3]; *(float4*)(p1)     = s;
    s.x = acc1[4]; s.y = acc1[5]; s.z = acc1[6]; s.w = acc1[7]; *(float4*)(p1 + 4) = s;
}

// ---------------------------------------------------------------------------
// reduce K-splits; optionally emit row-major f32 and/or transposed bf16 Xt.
// ---------------------------------------------------------------------------
__global__ void reduce_part_kernel(const float* __restrict__ part,
                                   float* __restrict__ outF,    // [M x 32] or null
                                   __bf16* __restrict__ outXt)  // [32 x KPAD] or null
{
    int idx = blockIdx.x * blockDim.x + threadIdx.x;
    if (idx >= 32 * KPAD) return;
    int m  = idx % KPAD;
    int nn = idx / KPAD;
    if (m < NN) {
        int tile = m >> 4, mi = m & 15;
        size_t base = (size_t)tile * 512 + (size_t)nn * 16 + mi;
        float s = 0.0f;
#pragma unroll
        for (int sp = 0; sp < NSPLIT; ++sp)
            s += part[(size_t)sp * NTILES * 512 + base];
        if (outF)  outF[(size_t)m * 32 + nn] = s;
        if (outXt) outXt[(size_t)nn * KPAD + m] = f2bf(s);
    } else {
        if (outXt) outXt[(size_t)nn * KPAD + m] = f2bf(0.0f);  // zero pad rows
    }
}

// f32 [M x 32] row-major -> bf16 Xt [32 x KPAD] (zero-padded)
__global__ void to_xt_kernel(const float* __restrict__ F, __bf16* __restrict__ Xt)
{
    int idx = blockIdx.x * blockDim.x + threadIdx.x;
    if (idx >= 32 * KPAD) return;
    int m = idx % KPAD;
    int n = idx / KPAD;
    Xt[(size_t)n * KPAD + m] = (m < NN) ? f2bf(F[(size_t)m * 32 + n]) : f2bf(0.0f);
}

// ---------------------------------------------------------------------------
// generic small dense layer: C = act(A[Mxlda(:K)] @ W[KxN] + b)
// act: 0=none 1=relu 2=tanh
// ---------------------------------------------------------------------------
__global__ __launch_bounds__(256) void small_gemm_kernel(
    const float* __restrict__ A, int lda,
    const float* __restrict__ W, const float* __restrict__ bias,
    float* __restrict__ C, int ldc,
    int M, int Nc, int K, int act)
{
    int tx = threadIdx.x;                 // 0..31 : column
    int ty = threadIdx.y;                 // 0..7  : row
    int m = blockIdx.x * 8 + ty;
    int n = blockIdx.y * 32 + tx;
    if (m >= M || n >= Nc) return;
    float acc = bias[n];
    const float* a = A + (size_t)m * lda;
    const float* w = W + n;
#pragma unroll 4
    for (int k = 0; k < K; ++k)
        acc = fmaf(a[k], w[(size_t)k * Nc], acc);
    if (act == 1)      acc = fmaxf(acc, 0.0f);
    else if (act == 2) acc = tanhf(acc);
    C[(size_t)m * ldc + n] = acc;
}

// copy a 32-wide column block (strided src -> strided dst)
__global__ void copy_block_kernel(const float* __restrict__ src, int lds,
                                  float* __restrict__ dst, int ldd)
{
    int idx = blockIdx.x * blockDim.x + threadIdx.x;
    if (idx >= NN * 32) return;
    int m = idx >> 5, c = idx & 31;
    dst[(size_t)m * ldd + c] = src[(size_t)m * lds + c];
}

// ---------------------------------------------------------------------------
// Host-side orchestration
// ---------------------------------------------------------------------------
static inline size_t alignup(size_t x) { return (x + 255) & ~(size_t)255; }

extern "C" void kernel_launch(void* const* d_in, const int* in_sizes, int n_in,
                              void* d_out, int out_size, void* d_ws, size_t ws_size,
                              hipStream_t stream) {
    (void)in_sizes; (void)n_in; (void)out_size; (void)ws_size;

    // --- inputs (JAX pytree flattening: dict keys sorted, {"w","b"} -> b,w) ---
    const float* adj       = (const float*)d_in[0];
    const float* adj_d     = (const float*)d_in[1];
    const float* adj_s     = (const float*)d_in[2];
    const float* node_feat = (const float*)d_in[3];
    const float* fc1_b  = (const float*)d_in[4];
    const float* fc1_w  = (const float*)d_in[5];
    const float* fc2_b  = (const float*)d_in[6];
    const float* fc2_w  = (const float*)d_in[7];
    const float* lat1_b = (const float*)d_in[8];
    const float* lat1_w = (const float*)d_in[9];
    const float* lat2_b = (const float*)d_in[10];
    const float* lat2_w = (const float*)d_in[11];
    // layers: base 12 + 16*l : mlp(b1,w1,b2,w2), phi(...), psi1(...), psi2(...)
    const float* nf1_b = (const float*)d_in[60];
    const float* nf1_w = (const float*)d_in[61];
    const float* nf2_b = (const float*)d_in[62];
    const float* nf2_w = (const float*)d_in[63];

    float* out_predict = (float*)d_out;            // [10000]
    float* out_latent  = (float*)d_out + NN;       // [10000 x 64]

    // --- workspace carve-up ---
    char* ws = (char*)d_ws;
    size_t off = 0;
    float* tmp256 = (float*)(ws + off); off += alignup((size_t)NN * 256 * 4);
    float* tmp512 = (float*)(ws + off); off += alignup((size_t)NN * 512 * 4);
    float* hcat   = (float*)(ws + off); off += alignup((size_t)NN * 128 * 4);
    float* cc     = (float*)(ws + off); off += alignup((size_t)NN * 96  * 4);
    float* t96    = (float*)(ws + off); off += alignup((size_t)NN * 96  * 4);
    float* phi32  = (float*)(ws + off); off += alignup((size_t)NN * 32  * 4);
    float* d32    = (float*)(ws + off); off += alignup((size_t)NN * 32  * 4);
    float* part   = (float*)(ws + off); off += alignup((size_t)NSPLIT * NTILES * 512 * 4);
    __bf16* xt_a  = (__bf16*)(ws + off); off += alignup((size_t)32 * KPAD * 2);
    __bf16* xt_b  = (__bf16*)(ws + off); off += alignup((size_t)32 * KPAD * 2);

    const dim3 blkSG(32, 8, 1);
    auto sg_grid = [](int M, int Nc) { return dim3((M + 7) / 8, (Nc + 31) / 32, 1); };
    const int elemXT = 32 * KPAD;
    const dim3 grdXT((elemXT + 255) / 256, 1, 1);
    const dim3 grdCP((NN * 32 + 255) / 256, 1, 1);
    const dim3 grdBG(NTILES, NSPLIT, 1);

    // --- node encoder: tanh(Lin 64->256) -> tanh(Lin 256->32) -> hcat[:,0:32]
    small_gemm_kernel<<<sg_grid(NN, 256), blkSG, 0, stream>>>(
        node_feat, 64, nf1_w, nf1_b, tmp256, 256, NN, 256, 64, 2);
    small_gemm_kernel<<<sg_grid(NN, 32), blkSG, 0, stream>>>(
        tmp256, 256, nf2_w, nf2_b, hcat, 128, NN, 32, 256, 2);

    // --- message passing layers ---
    for (int l = 0; l < 3; ++l) {
        const int b = 12 + 16 * l;
        const float* mlp1_b = (const float*)d_in[b + 0];
        const float* mlp1_w = (const float*)d_in[b + 1];
        const float* mlp2_b = (const float*)d_in[b + 2];
        const float* mlp2_w = (const float*)d_in[b + 3];
        const float* phi1_b = (const float*)d_in[b + 4];
        const float* phi1_w = (const float*)d_in[b + 5];
        const float* phi2_b = (const float*)d_in[b + 6];
        const float* phi2_w = (const float*)d_in[b + 7];
        const float* ps11_b = (const float*)d_in[b + 8];
        const float* ps11_w = (const float*)d_in[b + 9];
        const float* ps12_b = (const float*)d_in[b + 10];
        const float* ps12_w = (const float*)d_in[b + 11];
        const float* ps21_b = (const float*)d_in[b + 12];
        const float* ps21_w = (const float*)d_in[b + 13];
        const float* ps22_b = (const float*)d_in[b + 14];
        const float* ps22_w = (const float*)d_in[b + 15];
        const float* hid = hcat + (size_t)l * 32;     // [NN x 32] with lda=128

        // cc[:,0:32] = hid
        copy_block_kernel<<<grdCP, 256, 0, stream>>>(hid, 128, cc, 96);

        // phi MLP2 -> phi32 -> xt_a (bf16, transposed, padded)
        small_gemm_kernel<<<sg_grid(NN, 32), blkSG, 0, stream>>>(
            hid, 128, phi1_w, phi1_b, t96, 32, NN, 32, 32, 1);
        small_gemm_kernel<<<sg_grid(NN, 32), blkSG, 0, stream>>>(
            t96, 32, phi2_w, phi2_b, phi32, 32, NN, 32, 32, 0);
        to_xt_kernel<<<grdXT, 256, 0, stream>>>(phi32, xt_a);

        // net_agg = adj @ phi_out  -> xt_b (bf16 transposed)
        big_gemm_kernel<<<grdBG, 32, 0, stream>>>(adj, xt_a, part);
        reduce_part_kernel<<<grdXT, 256, 0, stream>>>(part, nullptr, xt_b);

        // h_drive = psi1(adj_d @ net_agg) -> cc[:,32:64]
        big_gemm_kernel<<<grdBG, 32, 0, stream>>>(adj_d, xt_b, part);
        reduce_part_kernel<<<grdXT, 256, 0, stream>>>(part, d32, nullptr);
        small_gemm_kernel<<<sg_grid(NN, 32), blkSG, 0, stream>>>(
            d32, 32, ps11_w, ps11_b, t96, 32, NN, 32, 32, 1);
        small_gemm_kernel<<<sg_grid(NN, 32), blkSG, 0, stream>>>(
            t96, 32, ps12_w, ps12_b, cc + 32, 96, NN, 32, 32, 0);

        // h_sink = psi2(adj_s @ net_agg) -> cc[:,64:96]
        big_gemm_kernel<<<grdBG, 32, 0, stream>>>(adj_s, xt_b, part);
        reduce_part_kernel<<<grdXT, 256, 0, stream>>>(part, d32, nullptr);
        small_gemm_kernel<<<sg_grid(NN, 32), blkSG, 0, stream>>>(
            d32, 32, ps21_w, ps21_b, t96, 32, NN, 32, 32, 1);
        small_gemm_kernel<<<sg_grid(NN, 32), blkSG, 0, stream>>>(
            t96, 32, ps22_w, ps22_b, cc + 64, 96, NN, 32, 32, 0);

        // hid_next = mlp(concat) -> hcat[:, (l+1)*32 : ]
        small_gemm_kernel<<<sg_grid(NN, 96), blkSG, 0, stream>>>(
            cc, 96, mlp1_w, mlp1_b, t96, 96, NN, 96, 96, 1);
        small_gemm_kernel<<<sg_grid(NN, 32), blkSG, 0, stream>>>(
            t96, 96, mlp2_w, mlp2_b, hcat + (size_t)(l + 1) * 32, 128, NN, 32, 96, 0);
    }

    // --- readout head ---
    small_gemm_kernel<<<sg_grid(NN, 256), blkSG, 0, stream>>>(
        hcat, 128, lat1_w, lat1_b, tmp256, 256, NN, 256, 128, 2);
    small_gemm_kernel<<<sg_grid(NN, 64), blkSG, 0, stream>>>(
        tmp256, 256, lat2_w, lat2_b, out_latent, 64, NN, 64, 256, 2);
    small_gemm_kernel<<<sg_grid(NN, 512), blkSG, 0, stream>>>(
        out_latent, 64, fc1_w, fc1_b, tmp512, 512, NN, 512, 64, 2);
    small_gemm_kernel<<<sg_grid(NN, 1), blkSG, 0, stream>>>(
        tmp512, 512, fc2_w, fc2_b, out_predict, 1, NN, 1, 512, 0);
}
